// Triplane_NeRF_111669150289
// MI455X (gfx1250) — compile-verified, matrix operand order checked by reading, not run on hardware
//
#include <hip/hip_runtime.h>
#include <math.h>

typedef _Float16 v16h __attribute__((ext_vector_type(16)));
typedef _Float16 v8h  __attribute__((ext_vector_type(8)));
typedef float    v8f  __attribute__((ext_vector_type(8)));

#define FEAT_RES 512
#define FEAT_DIM 48
#define WIDTH    128
#define PTS_PER_BLK 64
#define IN_PAD   160            // 138 padded to 5 * 32 (K-chunks of the f16 WMMA)

// workspace layout (units: halfs), all 16B-aligned
#define WT0_OFF 0               // Wt0[128][160]  (transposed, padded)
#define WT1_OFF 20480           // Wt1[128][128]
#define WT2_OFF 36864           // Wt2[128][128]
#define WT3_OFF 53248           // Wt3[16][128]   (4 real cols + 12 zero)
#define WS_HALFS 55296          // then floats: b0(128) b1(128) b2(128) b3p(16)

union V16 { v16h v; v8h h[2]; };

// ---------------- weight preprocessing: f32 -> padded transposed f16 ----------------
__global__ __launch_bounds__(256)
void prep_weights(const float* __restrict__ W0, const float* __restrict__ W1,
                  const float* __restrict__ W2, const float* __restrict__ W3,
                  const float* __restrict__ b0, const float* __restrict__ b1,
                  const float* __restrict__ b2, const float* __restrict__ b3,
                  _Float16* __restrict__ ws16, float* __restrict__ wsb)
{
    int t = blockIdx.x * 256 + threadIdx.x;
    if (t < 20480) {                               // Wt0[n][k] = W0[k][n], k<138 else 0
        int n = t / 160, k = t % 160;
        ws16[WT0_OFF + t] = (k < 138) ? (_Float16)W0[k * WIDTH + n] : (_Float16)0.0f;
    } else if (t < 36864) {                        // Wt1
        int u = t - 20480; int n = u / 128, k = u % 128;
        ws16[t] = (_Float16)W1[k * WIDTH + n];
    } else if (t < 53248) {                        // Wt2
        int u = t - 36864; int n = u / 128, k = u % 128;
        ws16[t] = (_Float16)W2[k * WIDTH + n];
    } else if (t < 55296) {                        // Wt3 (pad 4 -> 16 output cols)
        int u = t - 53248; int n = u / 128, k = u % 128;
        ws16[t] = (n < 4) ? (_Float16)W3[k * 4 + n] : (_Float16)0.0f;
    } else if (t < 55696) {                        // biases
        int u = t - 55296;
        float v;
        if      (u < 128) v = b0[u];
        else if (u < 256) v = b1[u - 128];
        else if (u < 384) v = b2[u - 256];
        else { int j = u - 384; v = (j < 4) ? b3[j] : 0.0f; }
        wsb[u] = v;
    }
}

// ---------------- one hidden MLP layer: LDS(f16) x global Wt(f16) -> LDS(f16), ReLU ----------------
template<int CH>
__device__ inline void mlp_layer(const _Float16* __restrict__ Wt, const float* __restrict__ bias,
                                 const _Float16* __restrict__ lin, int in_stride,
                                 _Float16* __restrict__ lout, int lane, int wv)
{
    const int nloc  = lane & 15;
    const int khalf = lane >> 4;
    const int ncol  = wv * 16 + nloc;             // this wave's output column

    // preload B fragments (reused across the 4 row tiles)
    V16 b[CH];
#pragma unroll
    for (int kc = 0; kc < CH; ++kc) {
        const _Float16* src = Wt + ncol * (CH * 32) + kc * 32 + khalf * 8;
        b[kc].h[0] = *(const v8h*)(src);
        b[kc].h[1] = *(const v8h*)(src + 16);
    }
    const float bv = bias[ncol];

#pragma unroll
    for (int r = 0; r < 4; ++r) {
        v8f acc = {0.f,0.f,0.f,0.f,0.f,0.f,0.f,0.f};
        const _Float16* arow = lin + (r * 16 + nloc) * in_stride + khalf * 8;
#pragma unroll
        for (int kc = 0; kc < CH; ++kc) {
            V16 a;
            a.h[0] = *(const v8h*)(arow + kc * 32);
            a.h[1] = *(const v8h*)(arow + kc * 32 + 16);
            acc = __builtin_amdgcn_wmma_f32_16x16x32_f16(false, a.v, false, b[kc].v,
                                                         (short)0, acc, false, false);
        }
#pragma unroll
        for (int j = 0; j < 8; ++j) {
            float v = fmaxf(acc[j] + bv, 0.0f);
            int m = r * 16 + j + khalf * 8;       // C/D layout: lanes<16 -> M=j, lanes>=16 -> M=j+8
            lout[m * WIDTH + ncol] = (_Float16)v;
        }
    }
}

// ---------------- final layer: 128 -> 4 (padded to 16), direct global store ----------------
__device__ inline void mlp_last(const _Float16* __restrict__ Wt3, const float* __restrict__ b3,
                                const _Float16* __restrict__ lin, float* __restrict__ out,
                                int pbase, int lane, int wv)
{
    if (wv >= 4) return;                          // wave-uniform: EXEC stays all-ones inside
    const int nloc  = lane & 15;
    const int khalf = lane >> 4;

    V16 b[4];
#pragma unroll
    for (int kc = 0; kc < 4; ++kc) {
        const _Float16* src = Wt3 + nloc * 128 + kc * 32 + khalf * 8;
        b[kc].h[0] = *(const v8h*)(src);
        b[kc].h[1] = *(const v8h*)(src + 16);
    }
    const float bv = b3[nloc];

    v8f acc = {0.f,0.f,0.f,0.f,0.f,0.f,0.f,0.f};
    const _Float16* arow = lin + (wv * 16 + nloc) * WIDTH + khalf * 8;
#pragma unroll
    for (int kc = 0; kc < 4; ++kc) {
        V16 a;
        a.h[0] = *(const v8h*)(arow + kc * 32);
        a.h[1] = *(const v8h*)(arow + kc * 32 + 16);
        acc = __builtin_amdgcn_wmma_f32_16x16x32_f16(false, a.v, false, b[kc].v,
                                                     (short)0, acc, false, false);
    }
    if (nloc < 4) {
#pragma unroll
        for (int j = 0; j < 8; ++j) {
            float v = acc[j] + bv;
            if (nloc == 3) v = fmaxf(v, 0.0f);    // relu only on density channel
            int m = pbase + wv * 16 + j + khalf * 8;
            out[m * 4 + nloc] = v;
        }
    }
}

// ---------------- fused kernel: encode + triplane sample + 4-layer WMMA MLP ----------------
__global__ __launch_bounds__(256)
void nerf_fused(const float* __restrict__ X, const float* __restrict__ D,
                const float* __restrict__ F, const _Float16* __restrict__ ws16,
                const float* __restrict__ wsb, float* __restrict__ out)
{
    __shared__ __align__(16) _Float16 bufA[PTS_PER_BLK * IN_PAD];   // 20 KB
    __shared__ __align__(16) _Float16 bufB[PTS_PER_BLK * WIDTH];    // 16 KB
    __shared__ int   sAddr[PTS_PER_BLK][3][4];                      // 3 KB
    __shared__ float sWgt [PTS_PER_BLK][3][4];                      // 3 KB

    const int tid  = threadIdx.x;
    const int lane = tid & 31;
    const int wv   = tid >> 5;

    // ---- Stage A: per-point positional encoding + bilinear setup (threads 0..63) ----
    if (tid < PTS_PER_BLK) {
        const int p  = tid;
        const int gp = blockIdx.x * PTS_PER_BLK + p;
        const int ray = gp >> 7;                  // N_SAMPLES = 128
        const float inv3 = 1.0f / 3.0f;
        float xn[3] = { X[gp*3+0]*inv3, X[gp*3+1]*inv3, X[gp*3+2]*inv3 };
        float dv[3] = { D[ray*3+0], D[ray*3+1], D[ray*3+2] };
        _Float16* hr = &bufA[p * IN_PAD];

        // PE(d, L=4): indices 48..74
        hr[48] = (_Float16)dv[0]; hr[49] = (_Float16)dv[1]; hr[50] = (_Float16)dv[2];
#pragma unroll
        for (int l = 0; l < 4; ++l) {
            float f = (float)(1 << l); int b = 51 + l * 6;
#pragma unroll
            for (int a = 0; a < 3; ++a) {
                float v = f * dv[a];
                hr[b + a]     = (_Float16)__sinf(v);
                hr[b + 3 + a] = (_Float16)__cosf(v);
            }
        }
        // PE(x, L=10): indices 75..137
        hr[75] = (_Float16)xn[0]; hr[76] = (_Float16)xn[1]; hr[77] = (_Float16)xn[2];
#pragma unroll
        for (int l = 0; l < 10; ++l) {
            float f = (float)(1 << l); int b = 78 + l * 6;
#pragma unroll
            for (int a = 0; a < 3; ++a) {
                float v = f * xn[a];
                hr[b + a]     = (_Float16)__sinf(v);
                hr[b + 3 + a] = (_Float16)__cosf(v);
            }
        }
#pragma unroll
        for (int k = 138; k < IN_PAD; ++k) hr[k] = (_Float16)0.0f;

        // bilinear corner addresses + weights for the 3 planes: (x,y) (y,z) (x,z)
        const float gxs[3] = { xn[0], xn[1], xn[0] };
        const float gys[3] = { xn[1], xn[2], xn[2] };
#pragma unroll
        for (int pl = 0; pl < 3; ++pl) {
            float ix = (gxs[pl] + 1.0f) * 0.5f * (float)(FEAT_RES - 1);
            float iy = (gys[pl] + 1.0f) * 0.5f * (float)(FEAT_RES - 1);
            float x0 = floorf(ix), y0 = floorf(iy);
            float wx1 = ix - x0, wx0 = 1.0f - wx1;
            float wy1 = iy - y0, wy0 = 1.0f - wy1;
            float cx[4] = { x0, x0 + 1.0f, x0,        x0 + 1.0f };
            float cy[4] = { y0, y0,        y0 + 1.0f, y0 + 1.0f };
            float cw[4] = { wx0*wy0, wx1*wy0, wx0*wy1, wx1*wy1 };
#pragma unroll
            for (int t = 0; t < 4; ++t) {
                bool valid = (cx[t] >= 0.0f) && (cx[t] <= 511.0f) &&
                             (cy[t] >= 0.0f) && (cy[t] <= 511.0f);
                int xc = (int)fminf(fmaxf(cx[t], 0.0f), 511.0f);
                int yc = (int)fminf(fmaxf(cy[t], 0.0f), 511.0f);
                sAddr[p][pl][t] = yc * FEAT_RES + xc;
                sWgt [p][pl][t] = valid ? cw[t] : 0.0f;
            }
        }
    }
    __syncthreads();

    // ---- Stage B: triplane gathers, (point,channel) tasks; 64*48 = 3072 = 12 * 256 ----
#pragma unroll 1
    for (int it = 0; it < 12; ++it) {
        int t = tid + it * 256;
        int p = t & 63;
        int c = t >> 6;                           // 0..47
        float sum = 0.0f;
#pragma unroll
        for (int pl = 0; pl < 3; ++pl) {
            const float* base = F + ((size_t)(pl * FEAT_DIM + c) << 18); // 512*512 = 2^18
#pragma unroll
            for (int tp = 0; tp < 4; ++tp)
                sum += base[sAddr[p][pl][tp]] * sWgt[p][pl][tp];
        }
        bufA[p * IN_PAD + c] = (_Float16)sum;
    }
    __syncthreads();

    // ---- MLP: WMMA chain ----
    mlp_layer<5>(ws16 + WT0_OFF, wsb + 0,   bufA, IN_PAD, bufB, lane, wv);
    __syncthreads();
    mlp_layer<4>(ws16 + WT1_OFF, wsb + 128, bufB, WIDTH,  bufA, lane, wv);
    __syncthreads();
    mlp_layer<4>(ws16 + WT2_OFF, wsb + 256, bufA, WIDTH,  bufB, lane, wv);
    __syncthreads();
    mlp_last(ws16 + WT3_OFF, wsb + 384, bufB, out, blockIdx.x * PTS_PER_BLK, lane, wv);
}

extern "C" void kernel_launch(void* const* d_in, const int* in_sizes, int n_in,
                              void* d_out, int out_size, void* d_ws, size_t ws_size,
                              hipStream_t stream)
{
    const float* X  = (const float*)d_in[0];
    const float* D  = (const float*)d_in[1];
    const float* F  = (const float*)d_in[2];
    const float* W0 = (const float*)d_in[3];
    const float* b0 = (const float*)d_in[4];
    const float* W1 = (const float*)d_in[5];
    const float* b1 = (const float*)d_in[6];
    const float* W2 = (const float*)d_in[7];
    const float* b2 = (const float*)d_in[8];
    const float* W3 = (const float*)d_in[9];
    const float* b3 = (const float*)d_in[10];

    _Float16* ws16 = (_Float16*)d_ws;
    float*    wsb  = (float*)((char*)d_ws + WS_HALFS * 2);

    prep_weights<<<(55696 + 255) / 256, 256, 0, stream>>>(W0, W1, W2, W3,
                                                          b0, b1, b2, b3, ws16, wsb);

    const int npts = 4096 * 128;
    nerf_fused<<<npts / PTS_PER_BLK, 256, 0, stream>>>(X, D, F, ws16, wsb, (float*)d_out);
}
